// babyLlama_78692390797834
// MI455X (gfx1250) — matrix operand order repaired and checked by reading, not compiled, not run on hardware
//
#include <hip/hip_runtime.h>
#include <math.h>

typedef _Float16 h16;
typedef __attribute__((ext_vector_type(16))) _Float16 v16h;
typedef __attribute__((ext_vector_type(8)))  _Float16 v8h;
typedef __attribute__((ext_vector_type(8)))  float    v8f;

#define DIM    1024
#define NHEADS 8
#define HD     128
#define HIDDEN 768
#define VOCAB  32000
#define BB     4
#define SS     1024
#define NTOK   (BB * SS)   // 4096

// ---------------------------------------------------------------------------
// WMMA helpers (CDNA5 16x16x32 f16 -> f32 accumulate)
// ---------------------------------------------------------------------------
__device__ __forceinline__ v8f wmma16(v16h a, v16h b, v8f c) {
  return __builtin_amdgcn_wmma_f32_16x16x32_f16(false, a, false, b, (short)0, c,
                                                false, false);
}

// A fragment (16xK tile, row-major src, leading dim `ld` in halves).
// Lane layout: lane<16 -> M=lane, K = {0..15}+k0
//              lane>=16 -> M=lane-16, K = {16..31}+k0
__device__ __forceinline__ v16h load_a_frag(const h16* src, int ld, int m0,
                                            int k0, int lane) {
  int m  = m0 + (lane & 15);
  int kb = k0 + ((lane & 16) ? 16 : 0);
  const h16* p = src + (size_t)m * ld + kb;
  v8h lo = *(const v8h*)(p);        // K = kb .. kb+7
  v8h hi = *(const v8h*)(p + 8);    // K = kb+8 .. kb+15
  v16h r;
#pragma unroll
  for (int i = 0; i < 8; ++i) { r[i] = lo[i]; r[8 + i] = hi[i]; }
  return r;
}

// B fragment (Kx16 tile) where the matrix is stored as W[N][Kdim] row-major
// (i.e. B[k][n] = W[n][k]). Lane layout: lane%16 = N, elements = 16
// consecutive K values, lanes>=16 take K+16. One contiguous 32-byte load.
__device__ __forceinline__ v16h load_b_frag(const h16* w, int ld, int n0,
                                            int k0, int lane) {
  int n  = n0 + (lane & 15);
  int kb = k0 + ((lane & 16) ? 16 : 0);
  return *(const v16h*)(w + (size_t)n * ld + kb);
}

// ---------------------------------------------------------------------------
// GEMM: C[M,N] = A[M,K] * W[N,K]^T (+ residual).  A,W f16; C f32.
// Block = 4 waves; each wave computes a 32x64 strip (2 A-frags x 4 B-frags =
// 8 WMMAs per 12 b128 loads per k-step); block tile = 128x64.
// ---------------------------------------------------------------------------
__global__ __launch_bounds__(128)
void gemm_f16f32(const h16* __restrict__ A, const h16* __restrict__ W,
                 const float* residual, float* C, int M, int N, int K) {
  int lane = threadIdx.x & 31;
  int wave = threadIdx.x >> 5;
  int m0 = blockIdx.x * 128 + wave * 32;
  int n0 = blockIdx.y * 64;
  v8f acc[2][4];
#pragma unroll
  for (int u = 0; u < 2; ++u)
#pragma unroll
    for (int t = 0; t < 4; ++t)
#pragma unroll
      for (int i = 0; i < 8; ++i) acc[u][t][i] = 0.f;

  // per-lane row pointers used for prefetch of the next k-tiles
  const h16* pfA = A + (size_t)(m0 + (lane & 15)) * K;
  const h16* pfW = W + (size_t)(n0 + 32 + (lane & 31)) * K;

  for (int k0 = 0; k0 < K; k0 += 32) {
    if (k0 + 64 < K) {
      __builtin_prefetch(pfA + k0 + 64, 0, 1);  // global_prefetch_b8
      __builtin_prefetch(pfW + k0 + 64, 0, 1);
    }
    v16h a0 = load_a_frag(A, K, m0,      k0, lane);
    v16h a1 = load_a_frag(A, K, m0 + 16, k0, lane);
#pragma unroll
    for (int t = 0; t < 4; ++t) {
      v16h b = load_b_frag(W, K, n0 + 16 * t, k0, lane);
      acc[0][t] = wmma16(a0, b, acc[0][t]);
      acc[1][t] = wmma16(a1, b, acc[1][t]);
    }
  }

  int rb  = (lane & 16) ? 8 : 0;
  int col = lane & 15;
#pragma unroll
  for (int u = 0; u < 2; ++u)
#pragma unroll
    for (int t = 0; t < 4; ++t)
#pragma unroll
      for (int r = 0; r < 8; ++r) {
        size_t idx = (size_t)(m0 + u * 16 + rb + r) * N +
                     (size_t)(n0 + 16 * t + col);
        float v = acc[u][t][r];
        if (residual) v += residual[idx];
        C[idx] = v;
      }
}

// ---------------------------------------------------------------------------
// Flash attention: one wave per (query-tile of 16, b*NHEADS+h).
// Q,K in [b*S, DIM] f16 (RoPE applied); V pre-transposed to [b,h,d,s] f16.
// ---------------------------------------------------------------------------
__global__ __launch_bounds__(32)
void attn_kernel(const h16* Q, const h16* Km, const h16* Vt, h16* O) {
  int lane = threadIdx.x;
  int q0 = blockIdx.x * 16;
  int bh = blockIdx.y;
  int b = bh >> 3, hh = bh & 7;
  const h16* Qb = Q  + (size_t)b * SS * DIM + hh * HD;
  const h16* Kb = Km + (size_t)b * SS * DIM + hh * HD;
  const h16* Vb = Vt + (size_t)bh * HD * SS;   // [HD][SS]

  v16h qa[4];
#pragma unroll
  for (int kk = 0; kk < 4; ++kk) qa[kk] = load_a_frag(Qb, DIM, q0, kk * 32, lane);

  v8f o[8];
  float mrow[8], lrow[8];
#pragma unroll
  for (int t = 0; t < 8; ++t) {
    mrow[t] = -1e30f; lrow[t] = 0.f;
#pragma unroll
    for (int i = 0; i < 8; ++i) o[t][i] = 0.f;
  }

  __shared__ __align__(32) h16 P[16 * 32];
  const float scale = 0.08838834764831845f;  // 1/sqrt(128)
  int rb = (lane & 16) ? 8 : 0;
  int c0 = lane & 15;

  for (int k0 = 0; k0 < q0 + 16; k0 += 32) {   // causal: skip fully-masked chunks
    v8f s0, s1;
#pragma unroll
    for (int i = 0; i < 8; ++i) { s0[i] = 0.f; s1[i] = 0.f; }
#pragma unroll
    for (int kk = 0; kk < 4; ++kk) {
      v16h b0 = load_b_frag(Kb, DIM, k0,      kk * 32, lane);
      v16h b1 = load_b_frag(Kb, DIM, k0 + 16, kk * 32, lane);
      s0 = wmma16(qa[kk], b0, s0);
      s1 = wmma16(qa[kk], b1, s1);
    }
#pragma unroll
    for (int r = 0; r < 8; ++r) {
      int qrow = q0 + rb + r;
      float a = s0[r] * scale + (((k0 + c0)      > qrow) ? -1e9f : 0.f);
      float c = s1[r] * scale + (((k0 + 16 + c0) > qrow) ? -1e9f : 0.f);
      float v = fmaxf(a, c);
#pragma unroll
      for (int off = 1; off < 16; off <<= 1) v = fmaxf(v, __shfl_xor(v, off, 32));
      float nm    = fmaxf(mrow[r], v);
      float alpha = __expf(mrow[r] - nm);
      mrow[r] = nm;
      a = __expf(a - nm);
      c = __expf(c - nm);
      float sum = a + c;
#pragma unroll
      for (int off = 1; off < 16; off <<= 1) sum += __shfl_xor(sum, off, 32);
      lrow[r] = lrow[r] * alpha + sum;
#pragma unroll
      for (int t = 0; t < 8; ++t) o[t][r] *= alpha;
      s0[r] = a; s1[r] = c;
    }
    __syncthreads();   // prior iteration's P reads done before overwrite
#pragma unroll
    for (int r = 0; r < 8; ++r) {
      P[(rb + r) * 32 + c0]      = (h16)s0[r];
      P[(rb + r) * 32 + 16 + c0] = (h16)s1[r];
    }
    __syncthreads();
    v16h pa = load_a_frag(P, 32, 0, 0, lane);
#pragma unroll
    for (int t = 0; t < 8; ++t) {
      v16h vb = load_b_frag(Vb, SS, t * 16, k0, lane);  // N=d, K=key
      o[t] = wmma16(pa, vb, o[t]);
    }
  }

#pragma unroll
  for (int t = 0; t < 8; ++t)
#pragma unroll
    for (int r = 0; r < 8; ++r) {
      size_t idx = ((size_t)b * SS + (size_t)(q0 + rb + r)) * DIM +
                   (size_t)(hh * HD + t * 16 + c0);
      O[idx] = (h16)(o[t][r] / lrow[r]);
    }
}

// ---------------------------------------------------------------------------
// Elementwise helpers
// ---------------------------------------------------------------------------
__global__ __launch_bounds__(256)
void embed_kernel(const int* tok, const float* emb, float* h) {
  int row = blockIdx.x;
  int t = tok[row];
  const float* e = emb + (size_t)t * DIM;
  float* o = h + (size_t)row * DIM;
  for (int i = threadIdx.x; i < DIM; i += 256) o[i] = e[i];
}

__global__ __launch_bounds__(256)
void rmsnorm_f16_kernel(const float* x, const float* w, h16* out, int dim) {
  int row = blockIdx.x;
  const float* xr = x + (size_t)row * dim;
  float ss = 0.f;
  for (int i = threadIdx.x; i < dim; i += 256) { float v = xr[i]; ss += v * v; }
#pragma unroll
  for (int off = 16; off > 0; off >>= 1) ss += __shfl_xor(ss, off, 32);
  __shared__ float part[8];
  if ((threadIdx.x & 31) == 0) part[threadIdx.x >> 5] = ss;
  __syncthreads();
  float tot = 0.f;
#pragma unroll
  for (int i = 0; i < 8; ++i) tot += part[i];
  float sc = rsqrtf(tot / (float)dim + 1e-6f);
  for (int i = threadIdx.x; i < dim; i += 256)
    out[(size_t)row * dim + i] = (h16)(xr[i] * sc * w[i]);
}

__global__ __launch_bounds__(256)
void f32_to_f16_kernel(const float* in, h16* out, size_t n) {
  size_t i = (size_t)blockIdx.x * 256 + threadIdx.x;
  if (i < n) out[i] = (h16)in[i];
}

__global__ __launch_bounds__(256)
void rope_f16_kernel(const float* x, h16* out) {
  size_t id = (size_t)blockIdx.x * 256 + threadIdx.x;
  if (id >= (size_t)NTOK * (DIM / 2)) return;
  size_t row = id / (DIM / 2);
  int p = (int)(id % (DIM / 2));
  int s = (int)(row & (SS - 1));
  int hp = p & (HD / 2 - 1);  // pair index within head: 0..63
  float inv = powf(10000.f, -(float)(2 * hp) / (float)HD);
  float ang = (float)s * inv;
  float cs = cosf(ang), sn = sinf(ang);
  size_t base = row * DIM + 2 * (size_t)p;
  float xr = x[base], xi = x[base + 1];
  out[base]     = (h16)(xr * cs - xi * sn);
  out[base + 1] = (h16)(xr * sn + xi * cs);
}

__global__ __launch_bounds__(256)
void vtrans_kernel(const float* v, h16* vt) {
  size_t id = (size_t)blockIdx.x * 256 + threadIdx.x;
  if (id >= (size_t)NTOK * DIM) return;
  size_t row = id / DIM;           // b*S + s
  int c = (int)(id % DIM);         // h*HD + d
  int b = (int)(row / SS), s = (int)(row % SS);
  int hh = c >> 7, d = c & 127;
  vt[(((size_t)b * NHEADS + hh) * HD + d) * SS + s] = (h16)v[id];
}

__global__ __launch_bounds__(256)
void silumul_kernel(const float* a1, const float* a3, h16* g, size_t n) {
  size_t i = (size_t)blockIdx.x * 256 + threadIdx.x;
  if (i < n) {
    float x = a1[i];
    float sig = 1.f / (1.f + __expf(-x));
    g[i] = (h16)(x * sig * a3[i]);
  }
}

// ---------------------------------------------------------------------------
// Orchestration
// ---------------------------------------------------------------------------
extern "C" void kernel_launch(void* const* d_in, const int* in_sizes, int n_in,
                              void* d_out, int out_size, void* d_ws, size_t ws_size,
                              hipStream_t stream) {
  (void)in_sizes; (void)n_in; (void)out_size; (void)ws_size;
  const int*   tokens     = (const int*)d_in[0];
  const float* tok_emb    = (const float*)d_in[1];
  const float* out_w      = (const float*)d_in[2];
  const float* final_norm = (const float*)d_in[3];

  const size_t MB = (size_t)1 << 20;
  char* p = (char*)d_ws;
  float* h    = (float*)p;  p += 16 * MB;   // fp32 residual stream [4096,1024]
  h16*  xn16  = (h16*)p;    p += 8  * MB;   // rmsnorm output, f16
  char* big   = p;                          // 64MB region, reused for out_w f16
  float* buf0 = (float*)p;  p += 16 * MB;   // q / ffn a1
  float* buf1 = (float*)p;  p += 16 * MB;   // k / ffn a3
  float* buf2 = (float*)p;  p += 16 * MB;   // v
  h16*  Q16   = (h16*)p;    p += 8  * MB;
  h16*  K16   = (h16*)p;    p += 8  * MB;
  h16*  Vt16  = (h16*)p;    p += 8  * MB;
  h16*  act16 = (h16*)p;    p += 8  * MB;   // attn out / ffn gate, f16
  h16*  wbuf  = (h16*)p;    p += 14 * MB;   // per-layer f16 weights
  h16*  outw16 = (h16*)big;                 // 62.5MB, overlays buf0..K16

  h16* wq16 = wbuf;
  h16* wk16 = wq16 + (size_t)DIM * DIM;
  h16* wv16 = wk16 + (size_t)DIM * DIM;
  h16* wo16 = wv16 + (size_t)DIM * DIM;
  h16* w116 = wo16 + (size_t)DIM * DIM;
  h16* w216 = w116 + (size_t)HIDDEN * DIM;
  h16* w316 = w216 + (size_t)DIM * HIDDEN;

  auto cvt = [&](const float* src, h16* dst, size_t n) {
    f32_to_f16_kernel<<<(unsigned)((n + 255) / 256), 256, 0, stream>>>(src, dst, n);
  };
  auto gemm = [&](const h16* A, const h16* W, const float* res, float* C,
                  int M, int N, int K) {
    dim3 g((unsigned)(M / 128), (unsigned)(N / 64));
    gemm_f16f32<<<g, 128, 0, stream>>>(A, W, res, C, M, N, K);
  };

  embed_kernel<<<NTOK, 256, 0, stream>>>(tokens, tok_emb, h);

  for (int l = 0; l < 4; ++l) {
    int base = 4 + l * 9;
    const float* wq = (const float*)d_in[base + 0];
    const float* wk = (const float*)d_in[base + 1];
    const float* wv = (const float*)d_in[base + 2];
    const float* wo = (const float*)d_in[base + 3];
    const float* w1 = (const float*)d_in[base + 4];
    const float* w2 = (const float*)d_in[base + 5];
    const float* w3 = (const float*)d_in[base + 6];
    const float* an = (const float*)d_in[base + 7];
    const float* fn = (const float*)d_in[base + 8];

    // attention block
    rmsnorm_f16_kernel<<<NTOK, 256, 0, stream>>>(h, an, xn16, DIM);
    cvt(wq, wq16, (size_t)DIM * DIM);
    cvt(wk, wk16, (size_t)DIM * DIM);
    cvt(wv, wv16, (size_t)DIM * DIM);
    cvt(wo, wo16, (size_t)DIM * DIM);
    cvt(w1, w116, (size_t)HIDDEN * DIM);
    cvt(w2, w216, (size_t)DIM * HIDDEN);
    cvt(w3, w316, (size_t)HIDDEN * DIM);

    gemm(xn16, wq16, nullptr, buf0, NTOK, DIM, DIM);
    gemm(xn16, wk16, nullptr, buf1, NTOK, DIM, DIM);
    gemm(xn16, wv16, nullptr, buf2, NTOK, DIM, DIM);

    {
      size_t n = (size_t)NTOK * (DIM / 2);
      rope_f16_kernel<<<(unsigned)((n + 255) / 256), 256, 0, stream>>>(buf0, Q16);
      rope_f16_kernel<<<(unsigned)((n + 255) / 256), 256, 0, stream>>>(buf1, K16);
    }
    {
      size_t n = (size_t)NTOK * DIM;
      vtrans_kernel<<<(unsigned)((n + 255) / 256), 256, 0, stream>>>(buf2, Vt16);
    }

    dim3 ag(SS / 16, BB * NHEADS);
    attn_kernel<<<ag, 32, 0, stream>>>(Q16, K16, Vt16, act16);

    gemm(act16, wo16, h, h, NTOK, DIM, DIM);   // h += attn @ wo^T

    // ffn block
    rmsnorm_f16_kernel<<<NTOK, 256, 0, stream>>>(h, fn, xn16, DIM);
    gemm(xn16, w116, nullptr, buf0, NTOK, HIDDEN, DIM);
    gemm(xn16, w316, nullptr, buf1, NTOK, HIDDEN, DIM);
    {
      size_t n = (size_t)NTOK * HIDDEN;
      silumul_kernel<<<(unsigned)((n + 255) / 256), 256, 0, stream>>>(buf0, buf1,
                                                                     act16, n);
    }
    gemm(act16, w216, h, h, NTOK, DIM, HIDDEN);  // h += g @ w2^T
  }

  // final norm + vocab projection
  rmsnorm_f16_kernel<<<NTOK, 256, 0, stream>>>(h, final_norm, xn16, DIM);
  cvt(out_w, outw16, (size_t)VOCAB * DIM);
  gemm(xn16, outw16, nullptr, (float*)d_out, NTOK, VOCAB, DIM);
}
